// FocalM_15650860826636
// MI455X (gfx1250) — matrix operand order repaired and checked by reading, not compiled, not run on hardware
//
#include <hip/hip_runtime.h>
#include <hip/hip_bf16.h>
#include <math.h>

#define B_ 8
#define C_ 192
#define H_ 96
#define W_ 96
#define HW_ (H_*W_)
#define M_ (B_*HW_)
#define LN_EPS_ 1e-5f

// Flip to 0 if the async-to-LDS inline asm fails to assemble.
#define USE_ASYNC_LDS 1

typedef __attribute__((ext_vector_type(16))) __bf16 bf16x16;
typedef __attribute__((ext_vector_type(8)))  __bf16 bf16x8;
typedef __attribute__((ext_vector_type(8)))  float  v8f;

__device__ __forceinline__ float gelu_f(float v) {
    return 0.5f * v * (1.0f + erff(v * 0.70710678118654752440f));
}
__device__ __forceinline__ float sigmoid_f(float v) {
    return 1.0f / (1.0f + __expf(-v));
}
// A-fragment K index for 16x32 bf16 A matrix (contiguous, matches B):
// lanes 0-15: halves 0..15 -> K 0..15
// lanes 16-31: halves 0..15 -> K 16..31
__device__ __forceinline__ int a_k(int lane, int j) {
    int hi = lane >> 4;
    return hi * 16 + j;
}
// Convert 16 contiguous floats -> bf16x16 (B-fragment rows are K-contiguous)
__device__ __forceinline__ bf16x16 cvt16(const float* __restrict__ p) {
    bf16x16 r;
    const float4* p4 = (const float4*)p;
#pragma unroll
    for (int i = 0; i < 4; ++i) {
        float4 v = p4[i];
        r[4*i+0] = (__bf16)v.x; r[4*i+1] = (__bf16)v.y;
        r[4*i+2] = (__bf16)v.z; r[4*i+3] = (__bf16)v.w;
    }
    return r;
}
// A-fragment from a K-contiguous fp32 row: 16 floats at k0+hi*16
__device__ __forceinline__ bf16x16 cvt_a_row(const float* __restrict__ row, int k0, int hi) {
    bf16x16 r;
    const float4* a = (const float4*)(row + k0 + hi * 16);
    const float4* b = (const float4*)(row + k0 + hi * 16 + 8);
#pragma unroll
    for (int i = 0; i < 2; ++i) {
        float4 v = a[i];
        r[4*i+0] = (__bf16)v.x; r[4*i+1] = (__bf16)v.y;
        r[4*i+2] = (__bf16)v.z; r[4*i+3] = (__bf16)v.w;
    }
#pragma unroll
    for (int i = 0; i < 2; ++i) {
        float4 v = b[i];
        r[8+4*i+0] = (__bf16)v.x; r[8+4*i+1] = (__bf16)v.y;
        r[8+4*i+2] = (__bf16)v.z; r[8+4*i+3] = (__bf16)v.w;
    }
    return r;
}

// ---------------------------------------------------------------------------
// 0) 3x3 avg pool of x (NCHW) written to out (NHWC); final GEMM accumulates on top.
__global__ __launch_bounds__(256) void avgpool_kernel(const float* __restrict__ x,
                                                      float* __restrict__ out)
{
    int p  = blockIdx.x * 256 + threadIdx.x;
    int bc = blockIdx.y;
    int b = bc / C_, c = bc % C_;
    int h = p / W_, w = p % W_;
    const float* plane = x + (size_t)bc * HW_;
    float s = 0.0f;
    for (int dy = -1; dy <= 1; ++dy) {
        int hh = h + dy; if (hh < 0 || hh >= H_) continue;
        for (int dx = -1; dx <= 1; ++dx) {
            int ww = w + dx; if (ww < 0 || ww >= W_) continue;
            s += plane[hh * W_ + ww];
        }
    }
    out[((size_t)b * HW_ + p) * C_ + c] = s * (1.0f / 9.0f);
}

// ---------------------------------------------------------------------------
// 1) xf GEMM: y[m,o] = sum_k x[k,m] * f_w[o,k] + f_b[o]
//    One block = 64 M-rows x ALL live N-tiles (12 ctx tiles + 1 gate tile),
//    so x is staged to LDS exactly once (async tensor-path loads).
__global__ __launch_bounds__(128) void gemm_xf_kernel(const float* __restrict__ x,
                                                      const float* __restrict__ f_w,
                                                      const float* __restrict__ f_b,
                                                      float* __restrict__ ctx,
                                                      float* __restrict__ gatesOut)
{
    const int t    = threadIdx.x;
    const int lane = t & 31;
    const int wv   = t >> 5;
    const int hi   = lane >> 4;
    const int lo   = lane & 15;
    const int m0   = blockIdx.x * 64;
    const int b    = m0 / HW_;
    const int p0   = m0 % HW_;

    __shared__ __attribute__((aligned(16))) float As[32][68];   // [k][m], padded
    v8f acc[13];
#pragma unroll
    for (int nt = 0; nt < 13; ++nt) acc[nt] = (v8f){};

    __builtin_prefetch(f_w, 0, 3);

    for (int k0 = 0; k0 < C_; k0 += 32) {
        __syncthreads();
        // stage 32(k) x 64(m) fp32 tile: 512 b128 transfers, 4 per thread
#pragma unroll
        for (int i = 0; i < 4; ++i) {
            int f  = t + i * 128;          // float4 id 0..511
            int kk = f >> 4;               // k row
            int ms = (f & 15) * 4;         // m offset
            const float* gp = x + (size_t)(b * C_ + k0 + kk) * HW_ + p0 + ms;
#if USE_ASYNC_LDS
            unsigned lds = (unsigned)(uintptr_t)&As[kk][ms];   // low 32 bits = LDS offset
            asm volatile("global_load_async_to_lds_b128 %0, %1, off"
                         :: "v"(lds), "v"(gp) : "memory");
#else
            *(float4*)&As[kk][ms] = *(const float4*)gp;
#endif
        }
#if USE_ASYNC_LDS
        asm volatile("s_wait_asynccnt 0x0" ::: "memory");
#endif
        __syncthreads();

        bf16x16 a;
#pragma unroll
        for (int j = 0; j < 16; ++j)
            a[j] = (__bf16)As[a_k(lane, j)][wv * 16 + lo];

#pragma unroll
        for (int nt = 0; nt < 13; ++nt) {
            const int o0 = (nt < 12) ? nt * 16 : 384;
            int o = o0 + lo;
            int o_safe = (o < 387) ? o : 386;       // dead cols never stored
            bf16x16 bm = cvt16(f_w + (size_t)o_safe * C_ + k0 + hi * 16);
            acc[nt] = __builtin_amdgcn_wmma_f32_16x16x32_bf16(false, a, false, bm,
                                                              (short)0, acc[nt], false, false);
        }
    }

#pragma unroll
    for (int nt = 0; nt < 13; ++nt) {
        const int o0 = (nt < 12) ? nt * 16 : 384;
        int o = o0 + lo;
        float bias = (o < 387) ? f_b[o] : 0.0f;
#pragma unroll
        for (int r = 0; r < 8; ++r) {
            int m = m0 + wv * 16 + hi * 8 + r;
            float v = acc[nt][r] + bias;
            if (nt < 12)      ctx[(size_t)m * C_ + o] = v;
            else if (lo < 3)  gatesOut[(size_t)m * 3 + lo] = v;
        }
    }
}

// ---------------------------------------------------------------------------
// 2) depthwise conv (groups=C) + GELU, NHWC. One block per pixel, one thread per channel.
template<int R>
__global__ __launch_bounds__(192) void dwconv_gelu_kernel(const float* __restrict__ in,
                                                          const float* __restrict__ wgt,
                                                          float* __restrict__ out)
{
    const int c = threadIdx.x;
    const int m = blockIdx.x;
    const int b = m / HW_;
    const int p = m % HW_;
    const int h = p / W_;
    const int w = p % W_;
    const int K = 2 * R + 1;
    float s = 0.0f;
    for (int dy = -R; dy <= R; ++dy) {
        int hh = h + dy; if (hh < 0 || hh >= H_) continue;
        for (int dx = -R; dx <= R; ++dx) {
            int ww = w + dx; if (ww < 0 || ww >= W_) continue;
            s += in[((size_t)(b * HW_ + hh * W_ + ww)) * C_ + c]
               * wgt[c * K * K + (dy + R) * K + (dx + R)];
        }
    }
    out[(size_t)m * C_ + c] = gelu_f(s);
}

// ---------------------------------------------------------------------------
// 3) single-channel gate conv + GELU -> ft[m]
template<int R>
__global__ __launch_bounds__(256) void gateconv_gelu_kernel(const float* __restrict__ gates,
                                                            int l,
                                                            const float* __restrict__ gw,
                                                            float* __restrict__ ft)
{
    int m = blockIdx.x * 256 + threadIdx.x;
    if (m >= M_) return;
    int b = m / HW_;
    int p = m % HW_;
    int h = p / W_, w = p % W_;
    const int K = 2 * R + 1;
    float s = 0.0f;
    for (int dy = -R; dy <= R; ++dy) {
        int hh = h + dy; if (hh < 0 || hh >= H_) continue;
        for (int dx = -R; dx <= R; ++dx) {
            int ww = w + dx; if (ww < 0 || ww >= W_) continue;
            s += gates[((size_t)(b * HW_ + hh * W_ + ww)) * 3 + l]
               * gw[(dy + R) * K + (dx + R)];
        }
    }
    ft[m] = gelu_f(s);
}

// ---------------------------------------------------------------------------
// 4) channel-LayerNorm of (ctx*ft) -> sigmoid -> gated accumulate into ctx_all.
//    One wave32 per pixel, shuffle reduction.
__global__ __launch_bounds__(256) void ln_sigma_accum_kernel(const float* __restrict__ ctx,
                                                             const float* __restrict__ ft,
                                                             const float* __restrict__ gates,
                                                             int l,
                                                             const float* __restrict__ g,
                                                             const float* __restrict__ bb,
                                                             float* __restrict__ ctx_all,
                                                             int accumulate)
{
    const int lane = threadIdx.x & 31;
    const int wv   = threadIdx.x >> 5;
    const int m    = blockIdx.x * 8 + wv;
    const float ftv = ft[m];
    float v[6], cx[6];
    float sum = 0.0f, sumsq = 0.0f;
#pragma unroll
    for (int i = 0; i < 6; ++i) {
        int c = lane + 32 * i;
        cx[i] = ctx[(size_t)m * C_ + c];
        v[i]  = cx[i] * ftv;
        sum   += v[i];
        sumsq += v[i] * v[i];
    }
#pragma unroll
    for (int off = 16; off > 0; off >>= 1) {
        sum   += __shfl_xor(sum,   off, 32);
        sumsq += __shfl_xor(sumsq, off, 32);
    }
    float mean = sum * (1.0f / C_);
    float var  = sumsq * (1.0f / C_) - mean * mean;
    float inv  = rsqrtf(var + LN_EPS_);
    float gate = gates[(size_t)m * 3 + l];
#pragma unroll
    for (int i = 0; i < 6; ++i) {
        int c = lane + 32 * i;
        float s = sigmoid_f((v[i] - mean) * inv * g[c] + bb[c]);
        float r = s * cx[i] + (1.0f - s) * gate;
        size_t idx = (size_t)m * C_ + c;
        ctx_all[idx] = accumulate ? (ctx_all[idx] + r) : r;
    }
}

// ---------------------------------------------------------------------------
// 5) spatial mean of ctx per (b,c)
__global__ void zero_kernel(float* p, int n)
{
    int i = blockIdx.x * 256 + threadIdx.x;
    if (i < n) p[i] = 0.0f;
}

__global__ __launch_bounds__(256) void ctx_mean_reduce_kernel(const float* __restrict__ ctx,
                                                              float* __restrict__ ctx_sum)
{
    __shared__ float acc[C_];
    int b  = blockIdx.x / 36;
    int p0 = (blockIdx.x % 36) * 256;
    for (int i = threadIdx.x; i < C_; i += 256) acc[i] = 0.0f;
    __syncthreads();
    for (int idx = threadIdx.x; idx < 256 * C_; idx += 256) {
        int p = idx / C_, c = idx % C_;
        atomicAdd(&acc[c], ctx[((size_t)b * HW_ + p0 + p) * C_ + c]);
    }
    __syncthreads();
    for (int i = threadIdx.x; i < C_; i += 256) atomicAdd(&ctx_sum[b * C_ + i], acc[i]);
}

// 5b) ctx_global = gelu(mean); fv = gelu(ctx_global * ctx_w1_center); per-batch fv stats
__global__ __launch_bounds__(192) void global_ctx_kernel(const float* __restrict__ ctx_sum,
                                                         const float* __restrict__ ctx_w1,
                                                         float* __restrict__ cg,
                                                         float* __restrict__ fv,
                                                         float* __restrict__ stats)
{
    int b = blockIdx.x;
    int c = threadIdx.x;
    __shared__ float s1[C_], s2[C_];
    float cgv = gelu_f(ctx_sum[b * C_ + c] * (1.0f / HW_));
    // 9x9 depthwise conv (pad 4) on 1x1 input: only center tap (4,4) survives
    float fvv = gelu_f(cgv * ctx_w1[c * 81 + 40]);
    cg[b * C_ + c] = cgv;
    fv[b * C_ + c] = fvv;
    s1[c] = fvv; s2[c] = fvv * fvv;
    __syncthreads();
    if (c < 64) { s1[c] += s1[c + 64] + s1[c + 128]; s2[c] += s2[c + 64] + s2[c + 128]; }
    __syncthreads();
    if (c < 32) { s1[c] += s1[c + 32]; s2[c] += s2[c + 32]; }
    __syncthreads();
    if (c == 0) {
        float sum = 0.0f, ss = 0.0f;
        for (int i = 0; i < 32; ++i) { sum += s1[i]; ss += s2[i]; }
        float mean = sum * (1.0f / C_);
        stats[b * 2]     = mean;
        stats[b * 2 + 1] = ss * (1.0f / C_) - mean * mean;
    }
}

// 6) global gating term: LN(fv*ftg) folded analytically, accumulate into ctx_all
__global__ __launch_bounds__(192) void global_term_kernel(const float* __restrict__ fv,
                                                          const float* __restrict__ cg,
                                                          const float* __restrict__ stats,
                                                          const float* __restrict__ ftg,
                                                          const float* __restrict__ gates,
                                                          const float* __restrict__ g,
                                                          const float* __restrict__ bb,
                                                          float* __restrict__ ctx_all)
{
    int m = blockIdx.x;
    int c = threadIdx.x;
    int b = m / HW_;
    float f    = ftg[m];
    float mean = stats[b * 2];
    float var  = stats[b * 2 + 1];
    float inv  = rsqrtf(f * f * var + LN_EPS_);
    float val  = (fv[b * C_ + c] - mean) * f * inv;
    float s    = sigmoid_f(val * g[c] + bb[c]);
    float gate = gates[(size_t)m * 3 + 2];
    ctx_all[(size_t)m * C_ + c] += s * cg[b * C_ + c] + (1.0f - s) * gate;
}

// ---------------------------------------------------------------------------
// 7) fused h_out GEMM -> *q -> proj GEMM -> += out (out pre-seeded with avgpool)
__global__ __launch_bounds__(128) void fused_proj_kernel(const float* __restrict__ ctx_all,
                                                         const float* __restrict__ q,
                                                         const float* __restrict__ h_w,
                                                         const float* __restrict__ h_b,
                                                         const float* __restrict__ proj_w,
                                                         const float* __restrict__ proj_b,
                                                         float* __restrict__ out)
{
    const int lane = threadIdx.x & 31;
    const int wv   = threadIdx.x >> 5;
    const int hi   = lane >> 4;
    const int lo   = lane & 15;
    const int m0   = blockIdx.x * 64;
    const int b    = m0 / HW_;
    const int p0   = m0 % HW_;

    __shared__ __attribute__((aligned(16))) __bf16 Hs[4][16][208]; // per-wave h-tile

    __builtin_prefetch(proj_w, 0, 3);

    const float* arow = ctx_all + (size_t)(m0 + wv * 16 + lo) * C_;

    // Hoisted stage-1 A fragments (reused for all 12 N-tiles)
    bf16x16 afrag[6];
#pragma unroll
    for (int kt = 0; kt < 6; ++kt)
        afrag[kt] = cvt_a_row(arow, kt * 32, hi);

    // Stage 1: h = ctx_all x h_w^T + h_b; then h *= q; stash bf16 tile in LDS
    for (int nt = 0; nt < 12; ++nt) {
        v8f acc = {};
#pragma unroll
        for (int kt = 0; kt < 6; ++kt) {
            bf16x16 bm = cvt16(h_w + (size_t)(nt * 16 + lo) * C_ + kt * 32 + hi * 16);
            acc = __builtin_amdgcn_wmma_f32_16x16x32_bf16(false, afrag[kt], false, bm,
                                                          (short)0, acc, false, false);
        }
        int n = nt * 16 + lo;
        float bias = h_b[n];
        const float* qcol = q + (size_t)(b * C_ + n) * HW_ + p0 + wv * 16 + hi * 8;
#pragma unroll
        for (int r = 0; r < 8; ++r) {
            float v = (acc[r] + bias) * qcol[r];
            Hs[wv][hi * 8 + r][n] = (__bf16)v;
        }
    }
    __syncthreads();

    // Hoisted stage-2 A fragments from LDS (aligned 16B runs, stride 208 halves)
    bf16x16 afrag2[6];
#pragma unroll
    for (int kt = 0; kt < 6; ++kt) {
        bf16x8 l8 = *(const bf16x8*)&Hs[wv][lo][kt * 32 + hi * 16];
        bf16x8 h8 = *(const bf16x8*)&Hs[wv][lo][kt * 32 + hi * 16 + 8];
        afrag2[kt] = __builtin_shufflevector(l8, h8, 0,1,2,3,4,5,6,7,8,9,10,11,12,13,14,15);
    }

    // Stage 2: out += (q*h) x proj_w^T + proj_b
    for (int nt = 0; nt < 12; ++nt) {
        v8f acc = {};
#pragma unroll
        for (int kt = 0; kt < 6; ++kt) {
            bf16x16 bm = cvt16(proj_w + (size_t)(nt * 16 + lo) * C_ + kt * 32 + hi * 16);
            acc = __builtin_amdgcn_wmma_f32_16x16x32_bf16(false, afrag2[kt], false, bm,
                                                          (short)0, acc, false, false);
        }
        int n = nt * 16 + lo;
        float bias = proj_b[n];
#pragma unroll
        for (int r = 0; r < 8; ++r) {
            size_t idx = (size_t)(m0 + wv * 16 + hi * 8 + r) * C_ + n;
            out[idx] += acc[r] + bias;
        }
    }
}

// ---------------------------------------------------------------------------
extern "C" void kernel_launch(void* const* d_in, const int* in_sizes, int n_in,
                              void* d_out, int out_size, void* d_ws, size_t ws_size,
                              hipStream_t stream)
{
    const float* x       = (const float*)d_in[0];
    const float* q       = (const float*)d_in[1];
    const float* f_w     = (const float*)d_in[2];
    const float* f_b     = (const float*)d_in[3];
    const float* h_w     = (const float*)d_in[4];
    const float* h_b     = (const float*)d_in[5];
    const float* ctx_w0  = (const float*)d_in[6];
    const float* ctx_w1  = (const float*)d_in[7];
    const float* gate_w0 = (const float*)d_in[8];
    const float* gate_w1 = (const float*)d_in[9];
    const float* ln_w    = (const float*)d_in[10];
    const float* ln_b    = (const float*)d_in[11];
    const float* proj_w  = (const float*)d_in[12];
    const float* proj_b  = (const float*)d_in[13];
    float* out = (float*)d_out;
    (void)in_sizes; (void)n_in; (void)out_size; (void)ws_size;

    float* ws = (float*)d_ws;
    const size_t MC = (size_t)M_ * C_;
    float* ctxA   = ws;  ws += MC;            // NHWC
    float* ctxB   = ws;  ws += MC;            // NHWC
    float* ctxall = ws;  ws += MC;            // NHWC
    float* gates  = ws;  ws += (size_t)M_ * 3;
    float* ft     = ws;  ws += M_;
    float* ftg    = ws;  ws += M_;
    float* csum   = ws;  ws += B_ * C_;
    float* cg     = ws;  ws += B_ * C_;
    float* fv     = ws;  ws += B_ * C_;
    float* stats  = ws;  ws += B_ * 2;

    // seed out with avgpool(x)
    avgpool_kernel<<<dim3(HW_ / 256, B_ * C_), 256, 0, stream>>>(x, out);
    // xf GEMM (x staged once, all live N-tiles per block)
    gemm_xf_kernel<<<M_ / 64, 128, 0, stream>>>(x, f_w, f_b, ctxA, gates);
    // level 0 (7x7)
    dwconv_gelu_kernel<3><<<M_, 192, 0, stream>>>(ctxA, ctx_w0, ctxB);
    gateconv_gelu_kernel<3><<<M_ / 256, 256, 0, stream>>>(gates, 0, gate_w0, ft);
    ln_sigma_accum_kernel<<<M_ / 8, 256, 0, stream>>>(ctxB, ft, gates, 0, ln_w, ln_b, ctxall, 0);
    // level 1 (9x9)
    dwconv_gelu_kernel<4><<<M_, 192, 0, stream>>>(ctxB, ctx_w1, ctxA);
    gateconv_gelu_kernel<4><<<M_ / 256, 256, 0, stream>>>(gates, 1, gate_w1, ft);
    ln_sigma_accum_kernel<<<M_ / 8, 256, 0, stream>>>(ctxA, ft, gates, 1, ln_w, ln_b, ctxall, 1);
    // global branch
    zero_kernel<<<(B_ * C_ + 255) / 256, 256, 0, stream>>>(csum, B_ * C_);
    ctx_mean_reduce_kernel<<<B_ * 36, 256, 0, stream>>>(ctxA, csum);
    global_ctx_kernel<<<B_, 192, 0, stream>>>(csum, ctx_w1, cg, fv, stats);
    gateconv_gelu_kernel<4><<<M_ / 256, 256, 0, stream>>>(gates, 2, gate_w1, ftg);
    global_term_kernel<<<M_, 192, 0, stream>>>(fv, cg, stats, ftg, gates, ln_w, ln_b, ctxall);
    // fused h_out GEMM -> *q -> proj GEMM -> += out
    fused_proj_kernel<<<M_ / 64, 128, 0, stream>>>(ctxall, q, h_w, h_b, proj_w, proj_b, out);
}
